// MoEGAT_45088566673466
// MI455X (gfx1250) — matrix operand order repaired and checked by reading, not compiled, not run on hardware
//
#include <hip/hip_runtime.h>
#include <hip/hip_bf16.h>

// ---- problem constants (match reference) ----
constexpr int BB = 8;     // batch
constexpr int NN = 1024;  // nodes
constexpr int DD = 512;   // hidden dim
constexpr int EE = 8;     // experts
constexpr float SLOPE = 0.2f;
constexpr float NEGV  = -1e9f;

// ---- CDNA5 WMMA types (bridge-doc verified) ----
typedef __attribute__((ext_vector_type(16))) __bf16 v16bf;
typedef __attribute__((ext_vector_type(8)))  float  v8f;

// Async global->LDS copy of one 16-byte chunk per lane (CDNA5 ASYNCcnt path).
// LDS address = low 32 bits of generic pointer (flat aperture maps addr[31:0]).
__device__ __forceinline__ void async_copy_b128(const __bf16* gsrc, __bf16* lds_dst) {
    uint32_t lds = (uint32_t)(uintptr_t)lds_dst;
    uint64_t gad = (uint64_t)(uintptr_t)gsrc;
    asm volatile("global_load_async_to_lds_b128 %0, %1, off"
                 :: "v"(lds), "v"(gad) : "memory");
}
__device__ __forceinline__ void async_wait0() {
    asm volatile("s_wait_asynccnt 0x0" ::: "memory");
}

// =====================================================================
// 1) cast x (fp32 -> bf16), row-major [B,N,D]
// =====================================================================
__global__ void cvt_x_kernel(const float* __restrict__ x, __bf16* __restrict__ xb, int total) {
    int i = blockIdx.x * blockDim.x + threadIdx.x;
    if (i < total) xb[i] = (__bf16)x[i];
}

// =====================================================================
// 2) transpose+cast W: W[e][d][z] fp32 -> Wt[e][z][d] bf16
// =====================================================================
__global__ void cvt_w_kernel(const float* __restrict__ W, __bf16* __restrict__ Wt, int total) {
    int i = blockIdx.x * blockDim.x + threadIdx.x;
    if (i >= total) return;
    int e   = i / (DD * DD);
    int rem = i % (DD * DD);
    int z   = rem / DD;
    int d   = rem % DD;
    Wt[i] = (__bf16)W[((size_t)e * DD + d) * DD + z];
}

// =====================================================================
// 3) gate = softmax(x @ gate_W + gate_b), one thread per (b,n)
// =====================================================================
__global__ void gate_kernel(const float* __restrict__ x, const float* __restrict__ gW,
                            const float* __restrict__ gb, float* __restrict__ gate, int total) {
    int i = blockIdx.x * blockDim.x + threadIdx.x;
    if (i >= total) return;
    const float* xr = x + (size_t)i * DD;
    float acc[EE];
#pragma unroll
    for (int e = 0; e < EE; ++e) acc[e] = gb[e];
    for (int d = 0; d < DD; ++d) {
        float xv = xr[d];
#pragma unroll
        for (int e = 0; e < EE; ++e) acc[e] += xv * gW[d * EE + e];
    }
    float m = acc[0];
#pragma unroll
    for (int e = 1; e < EE; ++e) m = fmaxf(m, acc[e]);
    float s = 0.f;
#pragma unroll
    for (int e = 0; e < EE; ++e) { acc[e] = __expf(acc[e] - m); s += acc[e]; }
    float inv = 1.0f / s;
#pragma unroll
    for (int e = 0; e < EE; ++e) gate[(size_t)i * EE + e] = acc[e] * inv;
}

// =====================================================================
// 4) hT[b,e,z,n] = (x[b] @ W[e])^T via v_wmma_f32_16x16x32_bf16
//    grid = (NN/16, BB*EE), block = 256 (8 waves x 4 col tiles each).
//    The shared 16x512 bf16 x-tile is staged into LDS once per block
//    with global_load_async_to_lds_b128 (ASYNCcnt), then all waves pull
//    A fragments from LDS; A is loaded once per k-step and reused for
//    4 back-to-back WMMAs.
// =====================================================================
__global__ __launch_bounds__(256) void hgemm_kernel(const __bf16* __restrict__ xb,
                                                    const __bf16* __restrict__ Wt,
                                                    __bf16* __restrict__ hT) {
    __shared__ __align__(16) __bf16 xtile[16 * DD];   // 16 KB

    const int be   = blockIdx.y;
    const int b    = be / EE;
    const int e    = be % EE;
    const int row0 = blockIdx.x * 16;                 // n tile

    // ---- async stage: 16 rows x 512 bf16 = 16 KB = 1024 x b128 chunks ----
    const int tid = threadIdx.x;
#pragma unroll
    for (int i = 0; i < 4; ++i) {
        int c   = tid + 256 * i;        // chunk id (16 B each)
        int row = c >> 6;               // 64 chunks per row
        int off = (c & 63) << 3;        // element offset (8 bf16 per chunk)
        async_copy_b128(xb + ((size_t)b * NN + row0 + row) * DD + off,
                        &xtile[row * DD + off]);
    }
    async_wait0();
    __syncthreads();

    const int wave = tid >> 5;
    const int lane = tid & 31;
    const int hw   = lane >> 4;
    const int l    = lane & 15;

    // each wave owns 4 column tiles: z cols [(wave*4+ct)*16 .. +15]
    const __bf16* wcol[4];
#pragma unroll
    for (int ct = 0; ct < 4; ++ct)
        wcol[ct] = Wt + ((size_t)e * DD + ((wave << 2) + ct) * 16 + l) * DD;

    v8f acc[4] = {};
    for (int k0 = 0; k0 < DD; k0 += 32) {
        union { v16bf v; uint4 q[2]; } A;
        // A from LDS: lane row = l, K-runs [k0+8*hw..+7], [k0+16+8*hw..+7]
        A.q[0] = *(const uint4*)&xtile[l * DD + k0 + 8 * hw];
        A.q[1] = *(const uint4*)&xtile[l * DD + k0 + 16 + 8 * hw];
#pragma unroll
        for (int ct = 0; ct < 4; ++ct) {
            union { v16bf v; uint4 q[2]; } Bf;
            const __bf16* bp = wcol[ct] + k0 + 8 * hw;
            Bf.q[0] = *(const uint4*)bp;
            Bf.q[1] = *(const uint4*)(bp + 16);
            acc[ct] = __builtin_amdgcn_wmma_f32_16x16x32_bf16(false, A.v, false, Bf.v,
                                                              (short)0, acc[ct], false, false);
        }
    }

    // C/D: lane col = l (z), rows M = 8*hw + v -> hT[z][n], v contiguous in n
#pragma unroll
    for (int ct = 0; ct < 4; ++ct) {
        const int colz = ((wave << 2) + ct) * 16 + l;
        __align__(16) __bf16 tmp[8];
#pragma unroll
        for (int v = 0; v < 8; ++v) tmp[v] = (__bf16)acc[ct][v];
        *(uint4*)&hT[((size_t)be * DD + colz) * NN + row0 + 8 * hw] = *(const uint4*)tmp;
    }
}

// =====================================================================
// 5) s_src/s_dst[b,e,n] = sum_z hT[b,e,z,n] * a_{src,dst}[e,z]
// =====================================================================
__global__ void svec_kernel(const __bf16* __restrict__ hT, const float* __restrict__ a_src,
                            const float* __restrict__ a_dst, float* __restrict__ ssrc,
                            float* __restrict__ sdst, int total) {
    int i = blockIdx.x * blockDim.x + threadIdx.x;
    if (i >= total) return;
    int be = i / NN;
    int n  = i % NN;
    int e  = be % EE;
    const __bf16* hp = hT + (size_t)be * DD * NN + n;
    float s1 = 0.f, s2 = 0.f;
    for (int z = 0; z < DD; ++z) {
        float hv = (float)hp[(size_t)z * NN];
        s1 += hv * a_src[(size_t)e * DD + z];
        s2 += hv * a_dst[(size_t)e * DD + z];
    }
    ssrc[i] = s1;
    sdst[i] = s2;
}

// =====================================================================
// 6) fused masked-softmax attention + P@H aggregation + ELU
//    grid = (NN/16, BB*EE), block = 256.
//    Softmax normalization folded into the accumulator; LDS holds only
//    unnormalized exp probs as bf16 (32 KB). Each wave loads A once per
//    k-step and issues 4 WMMAs into 4 accumulators.
// =====================================================================
__global__ __launch_bounds__(256) void attn_kernel(const __bf16* __restrict__ hT,
                                                   const float* __restrict__ ssrc,
                                                   const float* __restrict__ sdst,
                                                   const int* __restrict__ adj,
                                                   __bf16* __restrict__ out_e) {
    __shared__ __align__(16) __bf16 pb[16 * NN];   // 32 KB
    __shared__ float red[256];
    __shared__ float rowmax[16];
    __shared__ float rowinv[16];

    const int be = blockIdx.y;
    const int b  = be / EE;
    const int n0 = blockIdx.x * 16;

    const int tid = threadIdx.x;
    const int r   = tid >> 4;
    const int c   = tid & 15;

    const float* sd   = sdst + (size_t)be * NN;
    const float  rs   = ssrc[(size_t)be * NN + n0 + r];
    const int*   arow = adj + ((size_t)b * NN + n0 + r) * NN;

    // ---- pass 1: row max of masked LeakyReLU scores ----
    float m = -3.4e38f;
    for (int t = 0; t < NN / 16; ++t) {
        int j   = c + (t << 4);
        float v = rs + sd[j];
        v = (v > 0.f) ? v : v * SLOPE;
        v = (arow[j] > 0) ? v : NEGV;
        m = fmaxf(m, v);
    }
    red[tid] = m;
    __syncthreads();
    if (c == 0) {
        float mm = red[r * 16];
#pragma unroll
        for (int k = 1; k < 16; ++k) mm = fmaxf(mm, red[r * 16 + k]);
        rowmax[r] = mm;
    }
    __syncthreads();

    // ---- pass 2: exp(score - max) -> LDS bf16, accumulate row sum ----
    const float mm = rowmax[r];
    float s = 0.f;
    for (int t = 0; t < NN / 16; ++t) {
        int j   = c + (t << 4);
        float v = rs + sd[j];
        v = (v > 0.f) ? v : v * SLOPE;
        v = (arow[j] > 0) ? v : NEGV;
        float p = __expf(v - mm);
        s += p;
        pb[r * NN + j] = (__bf16)p;
    }
    red[tid] = s;
    __syncthreads();
    if (c == 0) {
        float ss = 0.f;
#pragma unroll
        for (int k = 0; k < 16; ++k) ss += red[r * 16 + k];
        rowinv[r] = 1.0f / ss;
    }
    __syncthreads();

    // ---- WMMA: out_tile[16, DD] = P(16xNN) @ H(NNxDD) ----
    const int wave = tid >> 5;
    const int lane = tid & 31;
    const int hw   = lane >> 4;
    const int l    = lane & 15;
    const __bf16* hTe = hT + (size_t)be * DD * NN;

    const __bf16* hcol[4];
#pragma unroll
    for (int ct = 0; ct < 4; ++ct)
        hcol[ct] = hTe + (size_t)(((wave << 2) + ct) * 16 + l) * NN;

    v8f acc[4] = {};
    for (int k0 = 0; k0 < NN; k0 += 32) {
        union { v16bf v; uint4 q[2]; } A;
        A.q[0] = *(const uint4*)&pb[l * NN + k0 + 8 * hw];
        A.q[1] = *(const uint4*)&pb[l * NN + k0 + 16 + 8 * hw];
#pragma unroll
        for (int ct = 0; ct < 4; ++ct) {
            union { v16bf v; uint4 q[2]; } Bf;
            const __bf16* bp = hcol[ct] + k0 + 8 * hw;
            Bf.q[0] = *(const uint4*)bp;
            Bf.q[1] = *(const uint4*)(bp + 16);
            acc[ct] = __builtin_amdgcn_wmma_f32_16x16x32_bf16(false, A.v, false, Bf.v,
                                                              (short)0, acc[ct], false, false);
        }
    }

    // fold softmax normalization, ELU, store bf16 [B,E,N,D]
#pragma unroll
    for (int ct = 0; ct < 4; ++ct) {
        const int colz = ((wave << 2) + ct) * 16 + l;
        const size_t base = (((size_t)be * NN) + n0 + 8 * hw) * DD + colz;
#pragma unroll
        for (int v = 0; v < 8; ++v) {
            float o = acc[ct][v] * rowinv[8 * hw + v];
            o = (o > 0.f) ? o : (__expf(o) - 1.0f);   // ELU (alpha=1)
            out_e[base + (size_t)v * DD] = (__bf16)o;
        }
    }
}

// =====================================================================
// 7) MoE combine: out[b,n,z] = sum_e out_e[b,e,n,z] * gate[b,n,e]
// =====================================================================
__global__ void combine_kernel(const __bf16* __restrict__ out_e, const float* __restrict__ gate,
                               float* __restrict__ out, int total) {
    int i = blockIdx.x * blockDim.x + threadIdx.x;
    if (i >= total) return;
    int b   = i / (NN * DD);
    int rem = i % (NN * DD);
    int n   = rem / DD;
    int z   = rem % DD;
    const float* g = gate + ((size_t)b * NN + n) * EE;
    float s = 0.f;
#pragma unroll
    for (int e = 0; e < EE; ++e)
        s += (float)out_e[(((size_t)b * EE + e) * NN + n) * DD + z] * g[e];
    out[i] = s;
}

// =====================================================================
// host-side launcher
// =====================================================================
extern "C" void kernel_launch(void* const* d_in, const int* in_sizes, int n_in,
                              void* d_out, int out_size, void* d_ws, size_t ws_size,
                              hipStream_t stream) {
    (void)in_sizes; (void)n_in; (void)out_size; (void)ws_size;

    const float* x      = (const float*)d_in[0];
    const int*   adj    = (const int*)  d_in[1];
    const float* gate_W = (const float*)d_in[2];
    const float* gate_b = (const float*)d_in[3];
    const float* W      = (const float*)d_in[4];
    const float* a_src  = (const float*)d_in[5];
    const float* a_dst  = (const float*)d_in[6];
    float* out          = (float*)d_out;

    char* p = (char*)d_ws;
    auto alloc = [&](size_t bytes) -> void* {
        void* r = (void*)p;
        p += (bytes + 255) & ~(size_t)255;
        return r;
    };
    __bf16* xb    = (__bf16*)alloc((size_t)BB * NN * DD * 2);        // 8 MB
    __bf16* Wt    = (__bf16*)alloc((size_t)EE * DD * DD * 2);        // 4 MB
    __bf16* hT    = (__bf16*)alloc((size_t)BB * EE * DD * NN * 2);   // 64 MB
    float*  ssrc  = (float*) alloc((size_t)BB * EE * NN * 4);
    float*  sdst  = (float*) alloc((size_t)BB * EE * NN * 4);
    float*  gate  = (float*) alloc((size_t)BB * NN * EE * 4);
    __bf16* out_e = (__bf16*)alloc((size_t)BB * EE * NN * DD * 2);   // 64 MB

    const int nxb = BB * NN * DD;
    cvt_x_kernel<<<(nxb + 255) / 256, 256, 0, stream>>>(x, xb, nxb);

    const int nwt = EE * DD * DD;
    cvt_w_kernel<<<(nwt + 255) / 256, 256, 0, stream>>>(W, Wt, nwt);

    const int nbn = BB * NN;
    gate_kernel<<<(nbn + 255) / 256, 256, 0, stream>>>(x, gate_W, gate_b, gate, nbn);

    dim3 hgrid(NN / 16, BB * EE);
    hgemm_kernel<<<hgrid, 256, 0, stream>>>(xb, Wt, hT);

    const int nsv = BB * EE * NN;
    svec_kernel<<<(nsv + 255) / 256, 256, 0, stream>>>(hT, a_src, a_dst, ssrc, sdst, nsv);

    dim3 agrid(NN / 16, BB * EE);
    attn_kernel<<<agrid, 256, 0, stream>>>(hT, ssrc, sdst, adj, out_e);

    const int nout = BB * NN * DD;
    combine_kernel<<<(nout + 255) / 256, 256, 0, stream>>>(out_e, gate, out, nout);
}
